// DDIGraphModel_7756710937203
// MI455X (gfx1250) — matrix operand order repaired and checked by reading, not compiled, not run on hardware
//
#include <hip/hip_runtime.h>

typedef __attribute__((ext_vector_type(16))) _Float16 v16h;
typedef __attribute__((ext_vector_type(8)))  _Float16 v8h;
typedef __attribute__((ext_vector_type(4)))  _Float16 v4h;
typedef __attribute__((ext_vector_type(8)))  float    v8f;

#define N_NODES 50000
#define N_EDGES 800000
#define N_QUERY 200000
#define EMB     128
#define HID     128
#define NCLS    86
#define NCLS_P  96   // padded to 6 tiles of 16

// ---------------------------------------------------------------------------
// One 16x16x32 f16 WMMA step.
//   a0 -> A[m][k0 + sel*8]       (8 contiguous halves)
//   a1 -> A[m][k0 + 16 + sel*8]  (8 contiguous halves)
//   b0 -> BT[n][k0 + sel*16]; loads BT[n][k0 + sel*8] and BT[n][k0 + 16 + sel*8] (8 halves each, BT = B transposed)
// ---------------------------------------------------------------------------
__device__ __forceinline__ v8f wmma_step(const _Float16* a0, const _Float16* a1,
                                         const _Float16* b0, v8f c) {
    v8h alo = *(const v8h*)a0;
    v8h ahi = *(const v8h*)a1;
    int bsel = (int)((threadIdx.x >> 4) & 1);
    v8h blo = *(const v8h*)(b0 - bsel * 8);
    v8h bhi = *(const v8h*)(b0 - bsel * 8 + 16);
    v16h a, b;
#pragma unroll
    for (int i = 0; i < 8; ++i) {
        a[i] = alo[i]; a[8 + i] = ahi[i];
        b[i] = blo[i]; b[8 + i] = bhi[i];
    }
    return __builtin_amdgcn_wmma_f32_16x16x32_f16(false, a, false, b,
                                                  (short)0, c, false, false);
}

// ---------------------------------------------------------------------------
// CSR construction
// ---------------------------------------------------------------------------
__global__ void k_zero_i32(int* p, int n) {
    int i = blockIdx.x * blockDim.x + threadIdx.x;
    if (i < n) p[i] = 0;
}

__global__ void k_count(const int* __restrict__ dst, int* cnt, int e_cnt) {
    int e = blockIdx.x * blockDim.x + threadIdx.x;
    if (e < e_cnt) atomicAdd(&cnt[dst[e]], 1);
}

// single-block chunked exclusive scan: row_start[i] = sum cnt[0..i), plus cursor copy
__global__ void k_scan(const int* __restrict__ cnt, int* __restrict__ row_start,
                       int* __restrict__ cursor, int n) {
    __shared__ int buf[256];
    __shared__ int carry;
    if (threadIdx.x == 0) carry = 0;
    __syncthreads();
    for (int base = 0; base < n; base += 256) {
        int i = base + (int)threadIdx.x;
        int v = (i < n) ? cnt[i] : 0;
        buf[threadIdx.x] = v;
        __syncthreads();
#pragma unroll
        for (int ofs = 1; ofs < 256; ofs <<= 1) {
            int t = (threadIdx.x >= (unsigned)ofs) ? buf[threadIdx.x - ofs] : 0;
            __syncthreads();
            buf[threadIdx.x] += t;
            __syncthreads();
        }
        int excl = buf[threadIdx.x] - v + carry;
        if (i < n) { row_start[i] = excl; cursor[i] = excl; }
        __syncthreads();
        if (threadIdx.x == 255) carry += buf[255];
        __syncthreads();
    }
    if (threadIdx.x == 0) row_start[n] = carry;
}

__global__ void k_dis(const int* __restrict__ cnt, float* __restrict__ dis, int n) {
    int i = blockIdx.x * blockDim.x + threadIdx.x;
    if (i < n) dis[i] = rsqrtf((float)cnt[i] + 1.0f);   // degree incl. self-loop
}

__global__ void k_fill_csr(const int* __restrict__ src, const int* __restrict__ dst,
                           int* __restrict__ cursor, int* __restrict__ csr_src, int e_cnt) {
    int e = blockIdx.x * blockDim.x + threadIdx.x;
    if (e >= e_cnt) return;
    int d = dst[e];
    int pos = atomicAdd(&cursor[d], 1);
    csr_src[pos] = src[e];
}

// ---------------------------------------------------------------------------
// f16 conversion / weight transpose prep
// ---------------------------------------------------------------------------
__global__ void k_emb2h(const int* __restrict__ x, const float* __restrict__ emb,
                        _Float16* __restrict__ out, int total) {
    int idx = blockIdx.x * blockDim.x + threadIdx.x;
    if (idx >= total) return;
    int i = idx >> 7, f = idx & 127;
    out[idx] = (_Float16)emb[x[i] * EMB + f];
}

// dst[n*K + k] = (n < N) ? src[k*N + n] : 0   (f32 -> f16 transpose w/ padding)
__global__ void k_tr(const float* __restrict__ src, _Float16* __restrict__ dst,
                     int K, int N, int Npad) {
    int idx = blockIdx.x * blockDim.x + threadIdx.x;
    if (idx >= Npad * K) return;
    int n = idx / K, k = idx - n * K;
    dst[idx] = (n < N) ? (_Float16)src[k * N + n] : (_Float16)0.0f;
}

// ---------------------------------------------------------------------------
// GEMM: out[M x 128] = A[M x 128](f16) @ B(via BT[128 x 128] f16), f32 out.
// grid.x = M/16 blocks, 256 threads = 8 waves, wave w -> n-tile w.
// ---------------------------------------------------------------------------
__global__ void k_gemm(const _Float16* __restrict__ A,
                       const _Float16* __restrict__ BT,
                       float* __restrict__ out) {
    int wave = threadIdx.x >> 5, lane = threadIdx.x & 31;
    int sel = lane >> 4, mn = lane & 15;
    int m0 = blockIdx.x * 16;
    int n0 = wave * 16;
    const _Float16* arow = A + (size_t)(m0 + mn) * HID;
    const _Float16* brow = BT + (size_t)(n0 + mn) * HID;
    v8f c = {};
#pragma unroll
    for (int k0 = 0; k0 < HID; k0 += 32)
        c = wmma_step(arow + k0 + sel * 8, arow + k0 + 16 + sel * 8,
                      brow + k0 + sel * 16, c);
    float* orow = out + (size_t)m0 * HID + n0;
#pragma unroll
    for (int r = 0; r < 8; ++r)
        orow[(size_t)(sel * 8 + r) * HID + mn] = c[r];
}

// ---------------------------------------------------------------------------
// Gather aggregation, fused with self-loop + bias (+ReLU) + f16 convert.
// One wave per node; lane owns a float4 feature slice (coalesced b128 loads).
// out_h[i] = act( sum_{e: dst=i} G[src_e]*dis[src_e]*dis[i]
//                 + G[i]*dis[i]^2 + bias )
// ---------------------------------------------------------------------------
__global__ void k_agg_gather(const float* __restrict__ G, const float* __restrict__ dis,
                             const int* __restrict__ row_start,
                             const int* __restrict__ csr_src,
                             const float* __restrict__ bias,
                             _Float16* __restrict__ out_h, int do_relu) {
    int wave = threadIdx.x >> 5, lane = threadIdx.x & 31;
    int node = blockIdx.x * 8 + wave;               // 50000 = 6250 * 8 exactly
    float di = dis[node];
    float4 g = *(const float4*)(G + (size_t)node * HID + lane * 4);
    float sl = di * di;
    float4 acc = make_float4(g.x * sl, g.y * sl, g.z * sl, g.w * sl);
    int beg = row_start[node], end = row_start[node + 1];
    for (int j = beg; j < end; ++j) {
        int s = csr_src[j];
        float norm = dis[s] * di;
        float4 v = *(const float4*)(G + (size_t)s * HID + lane * 4);
        acc.x += v.x * norm; acc.y += v.y * norm;
        acc.z += v.z * norm; acc.w += v.w * norm;
    }
    float4 b = *(const float4*)(bias + lane * 4);
    acc.x += b.x; acc.y += b.y; acc.z += b.z; acc.w += b.w;
    if (do_relu) {
        acc.x = fmaxf(acc.x, 0.0f); acc.y = fmaxf(acc.y, 0.0f);
        acc.z = fmaxf(acc.z, 0.0f); acc.w = fmaxf(acc.w, 0.0f);
    }
    v4h h; h[0] = (_Float16)acc.x; h[1] = (_Float16)acc.y;
    h[2] = (_Float16)acc.z; h[3] = (_Float16)acc.w;
    *(v4h*)(out_h + (size_t)node * HID + lane * 4) = h;
}

// ---------------------------------------------------------------------------
// Fused decoder: 16 queries / block. Phase 1: [z_s|z_d] @ dW1 + db1, ReLU ->
// LDS f16 tile. Phase 2 (waves 0..5, wave-uniform so EXEC all-ones for WMMA):
// hdec @ dW2 + db2 -> logits.
// ---------------------------------------------------------------------------
__global__ void k_decoder(const _Float16* __restrict__ Zh,
                          const int* __restrict__ qi,          // [2, Q] flat
                          const _Float16* __restrict__ dWT1,   // [128][256]
                          const float* __restrict__ db1,
                          const _Float16* __restrict__ dWT2,   // [96][128]
                          const float* __restrict__ db2,
                          float* __restrict__ out) {
    __shared__ _Float16 hdec[16 * HID];
    int wave = threadIdx.x >> 5, lane = threadIdx.x & 31;
    int sel = lane >> 4, mn = lane & 15;
    int q0 = blockIdx.x * 16;
    int q = q0 + mn;
    int s = qi[q], d = qi[N_QUERY + q];
    const _Float16* zs = Zh + (size_t)s * HID;
    const _Float16* zd = Zh + (size_t)d * HID;

    // ---- phase 1: hidden = relu([z_s, z_d] @ dW1 + db1) ----
    {
        int n2 = wave * 16 + mn;
        const _Float16* brow = dWT1 + (size_t)n2 * 256;
        v8f c = {};
#pragma unroll
        for (int k0 = 0; k0 < 256; k0 += 32) {
            const _Float16* arow = (k0 < 128) ? (zs + k0) : (zd + (k0 - 128));
            c = wmma_step(arow + sel * 8, arow + 16 + sel * 8,
                          brow + k0 + sel * 16, c);
        }
        float bias = db1[n2];
#pragma unroll
        for (int r = 0; r < 8; ++r) {
            float v = c[r] + bias;
            hdec[(sel * 8 + r) * HID + n2] = (_Float16)(v > 0.0f ? v : 0.0f);
        }
    }
    __syncthreads();

    // ---- phase 2: logits = hdec @ dW2 + db2 ----
    if (wave < 6) {                       // wave-uniform: EXEC stays all-ones
        const _Float16* arow = &hdec[mn * HID];
        int n2 = wave * 16 + mn;
        const _Float16* brow = dWT2 + (size_t)n2 * HID;
        v8f c = {};
#pragma unroll
        for (int k0 = 0; k0 < HID; k0 += 32)
            c = wmma_step(arow + k0 + sel * 8, arow + k0 + 16 + sel * 8,
                          brow + k0 + sel * 16, c);
        if (n2 < NCLS) {
            float bias = db2[n2];
#pragma unroll
            for (int r = 0; r < 8; ++r)
                out[(size_t)(q0 + sel * 8 + r) * NCLS + n2] = c[r] + bias;
        }
    }
}

// ---------------------------------------------------------------------------
// Launch
// ---------------------------------------------------------------------------
static inline size_t alignup(size_t x) { return (x + 255) & ~(size_t)255; }

extern "C" void kernel_launch(void* const* d_in, const int* in_sizes, int n_in,
                              void* d_out, int out_size, void* d_ws, size_t ws_size,
                              hipStream_t stream) {
    (void)in_sizes; (void)n_in; (void)out_size; (void)ws_size;

    const int*   x    = (const int*)d_in[0];
    const int*   ei   = (const int*)d_in[1];       // [2, E]
    const int*   eli  = (const int*)d_in[2];       // [2, Q]
    const float* emb  = (const float*)d_in[3];
    const float* W1   = (const float*)d_in[4];
    const float* b1   = (const float*)d_in[5];
    const float* W2   = (const float*)d_in[6];
    const float* b2   = (const float*)d_in[7];
    const float* dW1  = (const float*)d_in[8];
    const float* db1  = (const float*)d_in[9];
    const float* dW2  = (const float*)d_in[10];
    const float* db2  = (const float*)d_in[11];
    float* out = (float*)d_out;

    const int* e_src = ei;
    const int* e_dst = ei + N_EDGES;

    // ---- carve workspace (~55 MB) ----
    char* w = (char*)d_ws;
    size_t off = 0;
    auto carve = [&](size_t bytes) -> void* {
        void* p = w + off; off = alignup(off + bytes); return p;
    };
    int*      cnt   = (int*)     carve((size_t)N_NODES * 4);
    int*      rows  = (int*)     carve((size_t)(N_NODES + 1) * 4);
    int*      curs  = (int*)     carve((size_t)N_NODES * 4);
    int*      csr_s = (int*)     carve((size_t)N_EDGES * 4);
    float*    dis   = (float*)   carve((size_t)N_NODES * 4);
    _Float16* emb_h = (_Float16*)carve((size_t)N_NODES * EMB * 2);
    _Float16* WT1   = (_Float16*)carve((size_t)HID * HID * 2);
    _Float16* WT2   = (_Float16*)carve((size_t)HID * HID * 2);
    _Float16* dWT1  = (_Float16*)carve((size_t)HID * 256 * 2);
    _Float16* dWT2  = (_Float16*)carve((size_t)NCLS_P * HID * 2);
    float*    G     = (float*)   carve((size_t)N_NODES * HID * 4);
    _Float16* Hh    = (_Float16*)carve((size_t)N_NODES * HID * 2);

    const int T = 256;
    const int NF = N_NODES * HID;                  // 6.4M
    dim3 blk(T);

    // ---- CSR by destination + D^{-1/2} ----
    k_zero_i32<<<dim3((N_NODES + T - 1) / T), blk, 0, stream>>>(cnt, N_NODES);
    k_count   <<<dim3((N_EDGES + T - 1) / T), blk, 0, stream>>>(e_dst, cnt, N_EDGES);
    k_scan    <<<dim3(1), blk, 0, stream>>>(cnt, rows, curs, N_NODES);
    k_dis     <<<dim3((N_NODES + T - 1) / T), blk, 0, stream>>>(cnt, dis, N_NODES);
    k_fill_csr<<<dim3((N_EDGES + T - 1) / T), blk, 0, stream>>>(e_src, e_dst, curs, csr_s, N_EDGES);

    // ---- convert / transpose operands to f16 ----
    k_emb2h<<<dim3((NF + T - 1) / T), blk, 0, stream>>>(x, emb, emb_h, NF);
    k_tr<<<dim3((128 * 128 + T - 1) / T), blk, 0, stream>>>(W1,  WT1,  128, 128, 128);
    k_tr<<<dim3((128 * 128 + T - 1) / T), blk, 0, stream>>>(W2,  WT2,  128, 128, 128);
    k_tr<<<dim3((128 * 256 + T - 1) / T), blk, 0, stream>>>(dW1, dWT1, 256, 128, 128);
    k_tr<<<dim3((NCLS_P * 128 + T - 1) / T), blk, 0, stream>>>(dW2, dWT2, 128, NCLS, NCLS_P);

    // ---- layer 1: GEMM (WMMA) + fused gather-aggregate + ReLU -> f16 ----
    k_gemm<<<dim3(N_NODES / 16), blk, 0, stream>>>(emb_h, WT1, G);
    k_agg_gather<<<dim3(N_NODES / 8), blk, 0, stream>>>(G, dis, rows, csr_s, b1, Hh, 1);

    // ---- layer 2: GEMM (WMMA) + fused gather-aggregate -> Zh (f16) ----
    k_gemm<<<dim3(N_NODES / 16), blk, 0, stream>>>(Hh, WT2, G);
    k_agg_gather<<<dim3(N_NODES / 8), blk, 0, stream>>>(G, dis, rows, csr_s, b2, Hh, 0);

    // ---- fused decoder (WMMA x12) ----
    k_decoder<<<dim3(N_QUERY / 16), blk, 0, stream>>>(Hh, eli, dWT1, db1, dWT2, db2, out);
}
